// ResGatedGraphConvProcessor_28999619182970
// MI455X (gfx1250) — matrix operand order repaired and checked by reading, not compiled, not run on hardware
//
#include <hip/hip_runtime.h>

typedef __attribute__((ext_vector_type(16))) _Float16 v16h;
typedef __attribute__((ext_vector_type(8)))  _Float16 v8h;
typedef __attribute__((ext_vector_type(8)))  float    v8f;

#define LEAKY_SLOPE 0.01f
#define BN_EPS 1e-5f

// ---------------- weight conversion: W[l][n][k] (f32) -> Wt[l][mat][k][n] (f16) ----------
__global__ void conv_weights_kernel(const float* __restrict__ Wk,
                                    const float* __restrict__ Wq,
                                    const float* __restrict__ Wv,
                                    const float* __restrict__ Ws,
                                    _Float16* __restrict__ wt16,
                                    int nLayers)
{
    int idx = blockIdx.x * blockDim.x + threadIdx.x;
    int total = nLayers * 4 * 32 * 32;
    if (idx >= total) return;
    int n   = idx & 31;
    int k   = (idx >> 5) & 31;
    int mat = (idx >> 10) & 3;
    int l   = idx >> 12;
    const float* W;
    switch (mat) { case 0: W = Wk; break; case 1: W = Wq; break;
                   case 2: W = Wv; break; default: W = Ws; break; }
    wt16[((size_t)(l * 4 + mat) * 32 + k) * 32 + n] = (_Float16)W[l * 1024 + n * 32 + k];
}

__global__ void f32_to_f16_kernel(const float* __restrict__ in,
                                  _Float16* __restrict__ out, int n)
{
    int i = blockIdx.x * blockDim.x + threadIdx.x;
    if (i < n) out[i] = (_Float16)in[i];
}

__global__ void zero_stats_kernel(float* __restrict__ stats)
{
    stats[threadIdx.x] = 0.0f;
}

// ---------------- node projections via WMMA: k,q,v,skip = h @ W^T + b -----------------
// One wave handles a 16-row tile; 4 matrices x 2 column tiles = 8 v_wmma per wave.
__global__ void proj_wmma_kernel(const _Float16* __restrict__ hx,
                                 const _Float16* __restrict__ wt,  // [4][32][32], K-major
                                 const float* __restrict__ bk, const float* __restrict__ bq,
                                 const float* __restrict__ bv, const float* __restrict__ bs,
                                 float* __restrict__ kb, float* __restrict__ qb,
                                 float* __restrict__ vb, float* __restrict__ agg,
                                 int nRows, int nTiles)
{
    int wave = blockIdx.x * (blockDim.x >> 5) + (threadIdx.x >> 5);
    if (wave >= nTiles) return;
    int lane = threadIdx.x & 31;
    int m0   = wave * 16;
    int mrow = lane & 15;
    int loadRow = m0 + mrow;
    if (loadRow >= nRows) loadRow = nRows - 1;
    // A-matrix 16x32 f16 layout: lanes 0-15 hold K 0-7 / 16-23, lanes 16-31 hold K 8-15 / 24-31
    int ko = (lane < 16) ? 0 : 8;
    const _Float16* rp = hx + (size_t)loadRow * 32;
    v8h lo = *(const v8h*)(rp + ko);
    v8h hi = *(const v8h*)(rp + ko + 16);
    v16h a = __builtin_shufflevector(lo, hi, 0,1,2,3,4,5,6,7,8,9,10,11,12,13,14,15);

    int ncol = lane & 15;            // C/D: N = lane % 16
    int rofs = (lane < 16) ? 0 : 8;  // C/D: M = r (+8 for upper lanes)
    bool fullTile = (m0 + 16 <= nRows);   // wave-uniform -> single scalar branch
    size_t sbase = (size_t)(m0 + rofs) * 32 + ncol;
    float* const       outs[4] = { kb, qb, vb, agg };
    const float* const bia[4]  = { bk, bq, bv, bs };

#pragma unroll
    for (int mat = 0; mat < 4; ++mat) {
        const _Float16* wrow = wt + mat * 1024 + ko * 32 + ncol;  // B: lane holds column N=lane%16, K as A
#pragma unroll
        for (int nt = 0; nt < 2; ++nt) {
            int n0 = nt * 16;
            v16h b;
#pragma unroll
            for (int s = 0; s < 16; ++s) b[s] = wrow[((s < 8) ? s : s + 8) * 32 + n0];
            float bias = bia[mat][n0 + ncol];
            v8f c;
#pragma unroll
            for (int r = 0; r < 8; ++r) c[r] = bias;
            c = __builtin_amdgcn_wmma_f32_16x16x32_f16(false, a, false, b,
                                                       (short)0, c, false, false);
            float* op = outs[mat] + sbase + n0;   // per-lane base; rows at +r*32 floats
            if (fullTile) {
#pragma unroll
                for (int r = 0; r < 8; ++r) op[r * 32] = c[r];  // imm-offset store clause
            } else {
#pragma unroll
                for (int r = 0; r < 8; ++r)
                    if (m0 + rofs + r < nRows) op[r * 32] = c[r];
            }
        }
    }
}

// ---------------- edge stage: agg[dst] += sigmoid(k[dst]+q[src]) * v[src] ------------
// 8 lanes per edge, float4 per lane; all per-node arrays are L2-resident (12.8 MB each).
__global__ void edge_gate_kernel(const int* __restrict__ src, const int* __restrict__ dst,
                                 const float* __restrict__ kb, const float* __restrict__ qb,
                                 const float* __restrict__ vb, float* __restrict__ agg,
                                 int nEdges)
{
    long long tid = (long long)blockIdx.x * blockDim.x + threadIdx.x;
    int e = (int)(tid >> 3);
    if (e >= nEdges) return;
    int sub = (int)tid & 7;
    int c0  = sub * 4;
    if (sub == 0 && e + 8192 < nEdges) {
        __builtin_prefetch(src + e + 8192, 0, 0);
        __builtin_prefetch(dst + e + 8192, 0, 0);
    }
    int s = src[e];
    int d = dst[e];
    const float4 kd = *(const float4*)(kb + (size_t)d * 32 + c0);
    const float4 qs = *(const float4*)(qb + (size_t)s * 32 + c0);
    const float4 vs = *(const float4*)(vb + (size_t)s * 32 + c0);
    float* ap = agg + (size_t)d * 32 + c0;
    float g0 = 1.0f / (1.0f + __expf(-(kd.x + qs.x)));
    float g1 = 1.0f / (1.0f + __expf(-(kd.y + qs.y)));
    float g2 = 1.0f / (1.0f + __expf(-(kd.z + qs.z)));
    float g3 = 1.0f / (1.0f + __expf(-(kd.w + qs.w)));
    unsafeAtomicAdd(ap + 0, g0 * vs.x);
    unsafeAtomicAdd(ap + 1, g1 * vs.y);
    unsafeAtomicAdd(ap + 2, g2 * vs.z);
    unsafeAtomicAdd(ap + 3, g3 * vs.w);
}

// ---------------- z = leaky(conv) + h_prev (in place) + per-column sum/sumsq ----------
__global__ void bn_reduce_kernel(float* __restrict__ z, const float* __restrict__ hprev,
                                 float* __restrict__ stats, int nRows, int rowsPerBlock,
                                 int leaky)
{
    __shared__ float sh[512];
    int c = threadIdx.x & 31;
    int g = threadIdx.x >> 5;
    int base = blockIdx.x * rowsPerBlock;
    float s = 0.0f, ss = 0.0f;
    for (int r = g; r < rowsPerBlock; r += 8) {
        int row = base + r;
        if (row >= nRows) break;
        size_t i = (size_t)row * 32 + c;
        float t = z[i];
        if (leaky) t = (t >= 0.0f) ? t : LEAKY_SLOPE * t;
        float zz = t + hprev[i];
        z[i] = zz;
        s  += zz;
        ss += zz * zz;
    }
    sh[threadIdx.x]       = s;
    sh[256 + threadIdx.x] = ss;
    __syncthreads();
    if (threadIdx.x < 32) {
#pragma unroll
        for (int j = 1; j < 8; ++j) {
            s  += sh[j * 32 + c];
            ss += sh[256 + j * 32 + c];
        }
        unsafeAtomicAdd(stats + c, s);
        unsafeAtomicAdd(stats + 32 + c, ss);
    }
}

__global__ void bn_finalize_kernel(const float* __restrict__ stats,
                                   const float* __restrict__ gamma,
                                   const float* __restrict__ beta,
                                   float* __restrict__ scsh, int nRows)
{
    int t = threadIdx.x;
    if (t >= 32) return;
    float inv = 1.0f / (float)nRows;
    float mu  = stats[t] * inv;
    float var = stats[32 + t] * inv - mu * mu;
    float sc  = gamma[t] * rsqrtf(var + BN_EPS);
    scsh[t]      = sc;
    scsh[32 + t] = beta[t] - mu * sc;
}

__global__ void apply_kernel(const float* __restrict__ z, const float* __restrict__ scsh,
                             float* __restrict__ out, _Float16* __restrict__ hx16, int n)
{
    int i = blockIdx.x * blockDim.x + threadIdx.x;
    if (i >= n) return;
    int c = i & 31;
    float o = z[i] * scsh[c] + scsh[32 + c];
    out[i] = o;
    if (hx16) hx16[i] = (_Float16)o;
}

extern "C" void kernel_launch(void* const* d_in, const int* in_sizes, int n_in,
                              void* d_out, int out_size, void* d_ws, size_t ws_size,
                              hipStream_t stream)
{
    const float* x     = (const float*)d_in[0];
    const int*   ei    = (const int*)d_in[1];
    const float* Wk    = (const float*)d_in[3];
    const float* bk    = (const float*)d_in[4];
    const float* Wq    = (const float*)d_in[5];
    const float* bq    = (const float*)d_in[6];
    const float* Wv    = (const float*)d_in[7];
    const float* bv    = (const float*)d_in[8];
    const float* Ws    = (const float*)d_in[9];
    const float* bs    = (const float*)d_in[10];
    const float* gamma = (const float*)d_in[11];
    const float* beta  = (const float*)d_in[12];

    const int N = in_sizes[0] / 32;
    const int E = in_sizes[1] / 2;
    const int* src = ei;
    const int* dst = ei + E;

    size_t nd = (size_t)N * 32;
    float* kb  = (float*)d_ws;
    float* qb  = kb + nd;
    float* vb  = qb + nd;
    float* agg = vb + nd;
    float* hA  = agg + nd;
    _Float16* hx16 = (_Float16*)(hA + nd);
    _Float16* wt16 = hx16 + nd;                       // 3*4*32*32 halves
    float* stats = (float*)(wt16 + 3 * 4 * 32 * 32);  // 64 sums+sumsq
    float* scsh  = stats + 64;                        // 64 scale+shift

    int wTotal = 3 * 4 * 32 * 32;
    conv_weights_kernel<<<(wTotal + 255) / 256, 256, 0, stream>>>(Wk, Wq, Wv, Ws, wt16, 3);
    f32_to_f16_kernel<<<((int)nd + 255) / 256, 256, 0, stream>>>(x, hx16, (int)nd);

    int nTiles      = (N + 15) / 16;
    int projBlocks  = (nTiles + 7) / 8;
    long long eThreads = (long long)E * 8;
    int edgeBlocks  = (int)((eThreads + 255) / 256);
    int redBlocks   = 400;
    int rowsPerBlk  = (N + redBlocks - 1) / redBlocks;

    const float* hprev = x;
    for (int l = 0; l < 3; ++l) {
        proj_wmma_kernel<<<projBlocks, 256, 0, stream>>>(
            hx16, wt16 + (size_t)l * 4096,
            bk + l * 32, bq + l * 32, bv + l * 32, bs + l * 32,
            kb, qb, vb, agg, N, nTiles);
        zero_stats_kernel<<<1, 64, 0, stream>>>(stats);
        edge_gate_kernel<<<edgeBlocks, 256, 0, stream>>>(src, dst, kb, qb, vb, agg, E);
        bn_reduce_kernel<<<redBlocks, 256, 0, stream>>>(agg, hprev, stats, N, rowsPerBlk,
                                                        (l < 2) ? 1 : 0);
        bn_finalize_kernel<<<1, 32, 0, stream>>>(stats, gamma + l * 32, beta + l * 32,
                                                 scsh, N);
        float* hout = (l == 2) ? (float*)d_out : hA;
        apply_kernel<<<((int)nd + 255) / 256, 256, 0, stream>>>(
            agg, scsh, hout, (l < 2) ? hx16 : (_Float16*)nullptr, (int)nd);
        hprev = hA;
    }
    (void)n_in; (void)out_size; (void)ws_size;
}
